// AttnDecoderRNN_66030827209206
// MI455X (gfx1250) — compile-verified, matrix operand order checked by reading, not run on hardware
//
#include <hip/hip_runtime.h>
#include <cstdint>
#include <cstddef>

#define B_ 128
#define S_ 512
#define H_ 1024
#define H3_ 3072
#define NPRED_ 64
#define KIN_ 1027

typedef __attribute__((ext_vector_type(16))) __bf16 v16bf;
typedef __attribute__((ext_vector_type(8)))  float  v8f;

// ---------------------------------------------------------------------------
// 16-bit WMMA operand fragment: per ISA layout, each lane's 16 bf16 values are
// two contiguous runs of 8 elements (16 bytes each): [k0+8g .. +7] and
// [k0+16+8g .. +7].  Load as two b128s.
// ---------------------------------------------------------------------------
__device__ __forceinline__ v16bf ldfrag16(const __bf16* p) {
  union { v16bf v; uint4 q[2]; } u;
  u.q[0] = *reinterpret_cast<const uint4*>(p);
  u.q[1] = *reinterpret_cast<const uint4*>(p + 16);
  return u.v;
}

// C[M,N] = A[M,K](bf16,row-major) @ W[N,K]^T(bf16,row-major) + bias[N]
// One wave -> 16x64 tile (4 WMMA accumulators, shared A fragment).
// Software-pipelined with 2x-unrolled ping-pong buffers: each fragment buffer
// is written exactly once per iteration, so no rotation copies are generated
// and loads for one chunk overlap the WMMAs of the other.
// Requires K >= 64 and K % 64 == 0 (all call sites use K = 1024).
template <bool BF16OUT>
__global__ __launch_bounds__(256) void wmma_gemm_bf16(
    const __bf16* __restrict__ A, const __bf16* __restrict__ W,
    const float* __restrict__ bias, void* __restrict__ Cout,
    int M, int N, int K)
{
  const int wave = (int)((blockIdx.x * blockDim.x + threadIdx.x) >> 5);
  const int lane = (int)(threadIdx.x & 31);
  const int nGroups = N >> 6;     // N / 64
  const int mTiles  = M >> 4;     // M / 16
  if (wave >= mTiles * nGroups) return;   // wave-uniform: EXEC stays all-ones
  const int mT = wave / nGroups;
  const int nG = wave - mT * nGroups;
  const int g   = lane >> 4;      // K half-group
  const int sub = lane & 15;      // row (A) / col (B) within tile

  const __bf16* arow = A + (size_t)((mT << 4) + sub) * K + 8 * g;
  const __bf16* wrow[4];
#pragma unroll
  for (int t = 0; t < 4; ++t)
    wrow[t] = W + (size_t)((nG << 6) + (t << 4) + sub) * K + 8 * g;

  v8f zero = {};
  v8f acc[4];
#pragma unroll
  for (int t = 0; t < 4; ++t) acc[t] = zero;

  // prologue: chunk 0 -> buffer 0
  v16bf a0 = ldfrag16(arow);
  v16bf b0[4];
#pragma unroll
  for (int t = 0; t < 4; ++t) b0[t] = ldfrag16(wrow[t]);

  // steady state: two chunks per iteration, ping-pong buffers (no copies)
  int k = 0;
  for (int it = 0; it < (K >> 6) - 1; ++it) {
    __builtin_prefetch(arow + k + 512, 0, 3);
    // load chunk k+32 -> buffer 1
    v16bf a1 = ldfrag16(arow + k + 32);
    v16bf b1[4];
#pragma unroll
    for (int t = 0; t < 4; ++t) b1[t] = ldfrag16(wrow[t] + k + 32);
    // consume buffer 0 (chunk k)
#pragma unroll
    for (int t = 0; t < 4; ++t)
      acc[t] = __builtin_amdgcn_wmma_f32_16x16x32_bf16(
          false, a0, false, b0[t], (short)0, acc[t], false, false);
    // load chunk k+64 -> buffer 0
    a0 = ldfrag16(arow + k + 64);
#pragma unroll
    for (int t = 0; t < 4; ++t) b0[t] = ldfrag16(wrow[t] + k + 64);
    // consume buffer 1 (chunk k+32)
#pragma unroll
    for (int t = 0; t < 4; ++t)
      acc[t] = __builtin_amdgcn_wmma_f32_16x16x32_bf16(
          false, a1, false, b1[t], (short)0, acc[t], false, false);
    k += 64;
  }

  // epilogue: chunks K-64 (in buffer 0) and K-32
  {
    v16bf a1 = ldfrag16(arow + k + 32);
    v16bf b1[4];
#pragma unroll
    for (int t = 0; t < 4; ++t) b1[t] = ldfrag16(wrow[t] + k + 32);
#pragma unroll
    for (int t = 0; t < 4; ++t)
      acc[t] = __builtin_amdgcn_wmma_f32_16x16x32_bf16(
          false, a0, false, b0[t], (short)0, acc[t], false, false);
#pragma unroll
    for (int t = 0; t < 4; ++t)
      acc[t] = __builtin_amdgcn_wmma_f32_16x16x32_bf16(
          false, a1, false, b1[t], (short)0, acc[t], false, false);
  }

  // C/D layout: VGPR r -> M = r + 8*(lane>=16); N = lane&15
  const int rowBase = (mT << 4) + (g << 3);
#pragma unroll
  for (int t = 0; t < 4; ++t) {
    const int cc = (nG << 6) + (t << 4) + sub;
    const float bv = bias ? bias[cc] : 0.0f;
#pragma unroll
    for (int r = 0; r < 8; ++r) {
      const size_t idx = (size_t)(rowBase + r) * (size_t)N + cc;
      const float v = acc[t][r] + bv;
      if (BF16OUT) reinterpret_cast<__bf16*>(Cout)[idx] = (__bf16)v;
      else         reinterpret_cast<float*>(Cout)[idx]  = v;
    }
  }
}

// ---------------------------------------------------------------------------
// fp32 -> bf16 conversion, 4 elements/thread (n must be a multiple of 4)
// ---------------------------------------------------------------------------
__global__ void cvt4_f32_bf16(const float* __restrict__ in,
                              __bf16* __restrict__ outp, size_t n)
{
  size_t i = ((size_t)blockIdx.x * blockDim.x + threadIdx.x) * 4;
  if (i >= n) return;
  float4 v = *reinterpret_cast<const float4*>(in + i);
  outp[i + 0] = (__bf16)v.x;
  outp[i + 1] = (__bf16)v.y;
  outp[i + 2] = (__bf16)v.z;
  outp[i + 3] = (__bf16)v.w;
}

// Split W_ih(3H x 1027) into Wx fp32 (3H x 3) and Wc bf16 (3H x 1024)
__global__ void pack_wih(const float* __restrict__ W_ih,
                         float* __restrict__ Wx, __bf16* __restrict__ Wc)
{
  size_t idx = (size_t)blockIdx.x * blockDim.x + threadIdx.x;
  if (idx >= (size_t)H3_ * KIN_) return;
  int n = (int)(idx / KIN_);
  int k = (int)(idx - (size_t)n * KIN_);
  float v = W_ih[idx];
  if (k < 3) Wx[n * 3 + k] = v;
  else       Wc[(size_t)n * H_ + (k - 3)] = (__bf16)v;
}

// h = encoder_hidden[-1]; x = SOS(0)
__global__ void init_state(const float* __restrict__ ehid,
                           float* __restrict__ h, __bf16* __restrict__ hb,
                           float* __restrict__ x)
{
  int i = blockIdx.x * blockDim.x + threadIdx.x;
  if (i < B_ * H_) { float v = ehid[i]; h[i] = v; hb[i] = (__bf16)v; }
  if (i < B_ * 3) x[i] = 0.0f;
}

// scores[b,s] = Va . tanh(q[b,:] + UaK[b,s,:]) + Va_b   (one wave per (b,s))
__global__ void attn_scores(const float* __restrict__ q,
                            const __bf16* __restrict__ uak,
                            const float* __restrict__ va_w,
                            const float* __restrict__ va_b,
                            float* __restrict__ sc)
{
  int wave = (int)((blockIdx.x * blockDim.x + threadIdx.x) >> 5);
  int lane = (int)(threadIdx.x & 31);
  if (wave >= B_ * S_) return;
  int b = wave >> 9;
  int s = wave & (S_ - 1);
  const float*  qr = q   + (size_t)b * H_;
  const __bf16* ur = uak + ((size_t)b * S_ + s) * H_;
  float acc = 0.0f;
  for (int hh = lane; hh < H_; hh += 32)
    acc += tanhf(qr[hh] + (float)ur[hh]) * va_w[hh];
#pragma unroll
  for (int o = 16; o; o >>= 1) acc += __shfl_xor(acc, o, 32);
  if (lane == 0) sc[(size_t)b * S_ + s] = acc + va_b[0];
}

// softmax over S per batch row; also writes attentions output slice
__global__ void softmax_attn(const float* __restrict__ sc,
                             float* __restrict__ wout,
                             float* __restrict__ attn_out, int t)
{
  __shared__ float red[S_];
  int b = blockIdx.x, s = threadIdx.x;
  float v = sc[(size_t)b * S_ + s];
  red[s] = v; __syncthreads();
  for (int o = S_ / 2; o > 0; o >>= 1) {
    if (s < o) red[s] = fmaxf(red[s], red[s + o]);
    __syncthreads();
  }
  float m = red[0]; __syncthreads();
  float e = expf(v - m);
  red[s] = e; __syncthreads();
  for (int o = S_ / 2; o > 0; o >>= 1) {
    if (s < o) red[s] += red[s + o];
    __syncthreads();
  }
  float wv = e / red[0];
  wout[(size_t)b * S_ + s] = wv;
  attn_out[((size_t)b * NPRED_ + t) * S_ + s] = wv;
}

// ctx[b,h] = sum_s w[b,s] * enc[b,s,h]   (bf16 streaming, coalesced over h)
__global__ void ctx_kernel(const float* __restrict__ w,
                           const __bf16* __restrict__ encb,
                           float* __restrict__ ctx, __bf16* __restrict__ ctxb)
{
  int hh = blockIdx.x * blockDim.x + threadIdx.x;
  int b  = blockIdx.y;
  const float*  wr = w + (size_t)b * S_;
  const __bf16* ep = encb + (size_t)b * S_ * H_ + hh;
  float acc = 0.0f;
#pragma unroll 4
  for (int s = 0; s < S_; ++s) acc += wr[s] * (float)ep[(size_t)s * H_];
  ctx[(size_t)b * H_ + hh]  = acc;
  ctxb[(size_t)b * H_ + hh] = (__bf16)acc;
}

// GRU gate combine + tiny K=3 x-contribution; updates h (fp32 + bf16)
__global__ void gru_combine(const float* __restrict__ gi,
                            const float* __restrict__ gh,
                            const float* __restrict__ Wx,
                            const float* __restrict__ x,
                            float* __restrict__ h, __bf16* __restrict__ hb)
{
  int idx = blockIdx.x * blockDim.x + threadIdx.x;  // < B*H
  int b  = idx >> 10;
  int hh = idx & (H_ - 1);
  float x0 = x[b * 3 + 0], x1 = x[b * 3 + 1], x2 = x[b * 3 + 2];
  int j0 = hh, j1 = H_ + hh, j2 = 2 * H_ + hh;
  float gx0 = x0 * Wx[j0 * 3] + x1 * Wx[j0 * 3 + 1] + x2 * Wx[j0 * 3 + 2];
  float gx1 = x0 * Wx[j1 * 3] + x1 * Wx[j1 * 3 + 1] + x2 * Wx[j1 * 3 + 2];
  float gx2 = x0 * Wx[j2 * 3] + x1 * Wx[j2 * 3 + 1] + x2 * Wx[j2 * 3 + 2];
  size_t base = (size_t)b * H3_;
  float ir = gi[base + j0] + gx0, hr = gh[base + j0];
  float iz = gi[base + j1] + gx1, hz = gh[base + j1];
  float in_ = gi[base + j2] + gx2, hn = gh[base + j2];
  float r  = 1.0f / (1.0f + expf(-(ir + hr)));
  float zz = 1.0f / (1.0f + expf(-(iz + hz)));
  float nn = tanhf(in_ + r * hn);
  float ho = h[idx];
  float hnew = (1.0f - zz) * nn + zz * ho;
  h[idx]  = hnew;
  hb[idx] = (__bf16)hnew;
}

// pred = h_new @ out_w^T + out_b; raw pred -> x feedback; log_softmax -> out
__global__ void pred_logsoftmax(const float* __restrict__ h,
                                const float* __restrict__ out_w,
                                const float* __restrict__ out_b,
                                float* __restrict__ x,
                                float* __restrict__ preds, int t)
{
  int wave = (int)((blockIdx.x * blockDim.x + threadIdx.x) >> 5);
  int lane = (int)(threadIdx.x & 31);
  if (wave >= B_) return;
  const float* hr = h + (size_t)wave * H_;
  float a0 = 0.f, a1 = 0.f, a2 = 0.f;
  for (int k = lane; k < H_; k += 32) {
    float hv = hr[k];
    a0 += hv * out_w[k];
    a1 += hv * out_w[H_ + k];
    a2 += hv * out_w[2 * H_ + k];
  }
#pragma unroll
  for (int o = 16; o; o >>= 1) {
    a0 += __shfl_xor(a0, o, 32);
    a1 += __shfl_xor(a1, o, 32);
    a2 += __shfl_xor(a2, o, 32);
  }
  if (lane == 0) {
    float p0 = a0 + out_b[0], p1 = a1 + out_b[1], p2 = a2 + out_b[2];
    x[wave * 3 + 0] = p0; x[wave * 3 + 1] = p1; x[wave * 3 + 2] = p2;
    float m = fmaxf(p0, fmaxf(p1, p2));
    float lse = m + logf(expf(p0 - m) + expf(p1 - m) + expf(p2 - m));
    size_t bb = ((size_t)wave * NPRED_ + t) * 3;
    preds[bb + 0] = p0 - lse;
    preds[bb + 1] = p1 - lse;
    preds[bb + 2] = p2 - lse;
  }
}

__global__ void copy_hidden(const float* __restrict__ h, float* __restrict__ outp)
{
  int i = blockIdx.x * blockDim.x + threadIdx.x;
  if (i < B_ * H_) outp[i] = h[i];
}

// ---------------------------------------------------------------------------
extern "C" void kernel_launch(void* const* d_in, const int* in_sizes, int n_in,
                              void* d_out, int out_size, void* d_ws, size_t ws_size,
                              hipStream_t stream)
{
  (void)in_sizes; (void)n_in; (void)out_size; (void)ws_size;
  const float* enc   = (const float*)d_in[0];
  const float* ehid  = (const float*)d_in[1];
  const float* Wa_w  = (const float*)d_in[2];
  const float* Wa_b  = (const float*)d_in[3];
  const float* Ua_w  = (const float*)d_in[4];
  const float* Ua_b  = (const float*)d_in[5];
  const float* Va_w  = (const float*)d_in[6];
  const float* Va_b  = (const float*)d_in[7];
  const float* W_ih  = (const float*)d_in[8];
  const float* b_ih  = (const float*)d_in[9];
  const float* W_hh  = (const float*)d_in[10];
  const float* b_hh  = (const float*)d_in[11];
  const float* out_w = (const float*)d_in[12];
  const float* out_b = (const float*)d_in[13];

  float* out_f     = (float*)d_out;
  float* preds_out = out_f;                                  // (B, N, 3)
  float* hid_out   = out_f + (size_t)B_ * NPRED_ * 3;        // (1, B, H)
  float* attn_out  = hid_out + (size_t)B_ * H_;              // (B, N, S)

  char* ws = (char*)d_ws;
  size_t off = 0;
  auto carve = [&](size_t bytes) -> char* {
    char* p = ws + off;
    off = (off + bytes + 255) & ~(size_t)255;
    return p;
  };
  __bf16* enc_bf = (__bf16*)carve((size_t)B_ * S_ * H_ * 2);
  __bf16* uak_bf = (__bf16*)carve((size_t)B_ * S_ * H_ * 2);
  __bf16* wa_bf  = (__bf16*)carve((size_t)H_ * H_ * 2);
  __bf16* ua_bf  = (__bf16*)carve((size_t)H_ * H_ * 2);
  __bf16* whh_bf = (__bf16*)carve((size_t)H3_ * H_ * 2);
  __bf16* wc_bf  = (__bf16*)carve((size_t)H3_ * H_ * 2);
  float*  wx_f   = (float*)carve((size_t)H3_ * 3 * 4);
  float*  h_f    = (float*)carve((size_t)B_ * H_ * 4);
  __bf16* h_bf   = (__bf16*)carve((size_t)B_ * H_ * 2);
  float*  q_f    = (float*)carve((size_t)B_ * H_ * 4);
  float*  sc_f   = (float*)carve((size_t)B_ * S_ * 4);
  float*  w_f    = (float*)carve((size_t)B_ * S_ * 4);
  float*  ctx_f  = (float*)carve((size_t)B_ * H_ * 4);
  __bf16* ctx_bf = (__bf16*)carve((size_t)B_ * H_ * 2);
  float*  gi_f   = (float*)carve((size_t)B_ * H3_ * 4);
  float*  gh_f   = (float*)carve((size_t)B_ * H3_ * 4);
  float*  x_f    = (float*)carve((size_t)B_ * 3 * 4);

  // ---- one-time prep: bf16 conversions, W_ih split, state init, UaK GEMM
  const size_t nEnc = (size_t)B_ * S_ * H_;
  cvt4_f32_bf16<<<(unsigned)(nEnc / 4 / 256), 256, 0, stream>>>(enc, enc_bf, nEnc);
  cvt4_f32_bf16<<<(unsigned)((size_t)H_ * H_ / 4 / 256), 256, 0, stream>>>(Wa_w, wa_bf, (size_t)H_ * H_);
  cvt4_f32_bf16<<<(unsigned)((size_t)H_ * H_ / 4 / 256), 256, 0, stream>>>(Ua_w, ua_bf, (size_t)H_ * H_);
  cvt4_f32_bf16<<<(unsigned)((size_t)H3_ * H_ / 4 / 256), 256, 0, stream>>>(W_hh, whh_bf, (size_t)H3_ * H_);
  pack_wih<<<(unsigned)(((size_t)H3_ * KIN_ + 255) / 256), 256, 0, stream>>>(W_ih, wx_f, wc_bf);
  init_state<<<512, 256, 0, stream>>>(ehid, h_f, h_bf, x_f);

  {
    unsigned waves = (unsigned)((B_ * S_ / 16) * (H_ / 64));
    wmma_gemm_bf16<true><<<(waves + 7) / 8, 256, 0, stream>>>(
        enc_bf, ua_bf, Ua_b, uak_bf, B_ * S_, H_, H_);
  }

  // ---- 64 autoregressive steps
  const unsigned gemmQBlocks  = ((B_ / 16) * (H_ / 64) + 7) / 8;
  const unsigned gemm3Blocks  = ((B_ / 16) * (H3_ / 64) + 7) / 8;
  for (int t = 0; t < NPRED_; ++t) {
    wmma_gemm_bf16<false><<<gemmQBlocks, 256, 0, stream>>>(
        h_bf, wa_bf, Wa_b, q_f, B_, H_, H_);
    attn_scores<<<(unsigned)((size_t)B_ * S_ * 32 / 256), 256, 0, stream>>>(
        q_f, uak_bf, Va_w, Va_b, sc_f);
    softmax_attn<<<B_, S_, 0, stream>>>(sc_f, w_f, attn_out, t);
    ctx_kernel<<<dim3(H_ / 256, B_), 256, 0, stream>>>(w_f, enc_bf, ctx_f, ctx_bf);
    wmma_gemm_bf16<false><<<gemm3Blocks, 256, 0, stream>>>(
        ctx_bf, wc_bf, b_ih, gi_f, B_, H3_, H_);
    wmma_gemm_bf16<false><<<gemm3Blocks, 256, 0, stream>>>(
        h_bf, whh_bf, b_hh, gh_f, B_, H3_, H_);
    gru_combine<<<B_ * H_ / 256, 256, 0, stream>>>(gi_f, gh_f, wx_f, x_f, h_f, h_bf);
    pred_logsoftmax<<<(B_ * 32 + 255) / 256, 256, 0, stream>>>(
        h_f, out_w, out_b, x_f, preds_out, t);
  }

  copy_hidden<<<(B_ * H_ + 255) / 256, 256, 0, stream>>>(h_f, hid_out);
}
